// SelfAttention_35905926595225
// MI455X (gfx1250) — matrix operand order repaired and checked by reading, not compiled, not run on hardware
//
#include <hip/hip_runtime.h>

#define D_MODEL 1024
#define N_HEADS 16
#define D_K     64
#define BATCH   2
#define SEQ     2048

typedef __bf16 bf16_t;
typedef __bf16 v16bf __attribute__((ext_vector_type(16)));
typedef float  v8f   __attribute__((ext_vector_type(8)));

struct alignas(16) U4 { unsigned x, y, z, w; };
struct alignas(8)  U2 { unsigned x, y; };
union FragU { v16bf v; U4 q[2]; };

// Build a 16x bf16 WMMA operand from two 8-element (16B) chunks.
static __device__ inline v16bf load_frag2(const bf16_t* p0, const bf16_t* p1) {
    FragU f;
    f.q[0] = *reinterpret_cast<const U4*>(p0);
    f.q[1] = *reinterpret_cast<const U4*>(p1);
    return f.v;
}

// Generic LDS pointer -> 32-bit LDS byte offset (flat->LDS uses addr[31:0]).
static __device__ inline unsigned lds_addr32(const void* p) {
    return (unsigned)(size_t)p;
}

// CDNA5 async copy: 16 bytes per lane, global -> LDS, tracked with ASYNCcnt.
static __device__ inline void async_ld_b128(unsigned ldsoff, const void* gaddr) {
    asm volatile("global_load_async_to_lds_b128 %0, %1, off"
                 :: "v"(ldsoff), "v"(gaddr)
                 : "memory");
}
static __device__ inline void wait_async0() {
    asm volatile("s_wait_asynccnt 0x0" ::: "memory");
}

static __device__ inline float redmax16(float v) {
    v = fmaxf(v, __shfl_xor(v, 1));
    v = fmaxf(v, __shfl_xor(v, 2));
    v = fmaxf(v, __shfl_xor(v, 4));
    v = fmaxf(v, __shfl_xor(v, 8));
    return v;
}
static __device__ inline float redsum16(float v) {
    v += __shfl_xor(v, 1);
    v += __shfl_xor(v, 2);
    v += __shfl_xor(v, 4);
    v += __shfl_xor(v, 8);
    return v;
}

// ---------------- LayerNorm -> bf16 (one row per block, float4 per thread) ----------------
__global__ __launch_bounds__(256)
void ln_bf16_kernel(const float* __restrict__ x, const float* __restrict__ w,
                    const float* __restrict__ b, bf16_t* __restrict__ out) {
    const int row = blockIdx.x;
    const float* xr = x + (size_t)row * D_MODEL;
    const int i4 = threadIdx.x * 4;
    const float4 xv = *reinterpret_cast<const float4*>(xr + i4);
    float s  = xv.x + xv.y + xv.z + xv.w;
    float ss = xv.x * xv.x + xv.y * xv.y + xv.z * xv.z + xv.w * xv.w;
    #pragma unroll
    for (int off = 1; off < 32; off <<= 1) {
        s  += __shfl_xor(s, off);
        ss += __shfl_xor(ss, off);
    }
    __shared__ float rs[8], rss[8];
    const int wave = threadIdx.x >> 5, lane = threadIdx.x & 31;
    if (lane == 0) { rs[wave] = s; rss[wave] = ss; }
    __syncthreads();
    s = 0.f; ss = 0.f;
    #pragma unroll
    for (int i = 0; i < 8; ++i) { s += rs[i]; ss += rss[i]; }
    const float mu   = s * (1.f / D_MODEL);
    const float var  = ss * (1.f / D_MODEL) - mu * mu;
    const float rstd = rsqrtf(var + 1e-5f);
    const float4 wv = *reinterpret_cast<const float4*>(w + i4);
    const float4 bv = *reinterpret_cast<const float4*>(b + i4);
    union { U2 u; bf16_t e[4]; } pk;
    pk.e[0] = (bf16_t)((xv.x - mu) * rstd * wv.x + bv.x);
    pk.e[1] = (bf16_t)((xv.y - mu) * rstd * wv.y + bv.y);
    pk.e[2] = (bf16_t)((xv.z - mu) * rstd * wv.z + bv.z);
    pk.e[3] = (bf16_t)((xv.w - mu) * rstd * wv.w + bv.w);
    *reinterpret_cast<U2*>(out + (size_t)row * D_MODEL + i4) = pk.u;
}

// ---------------- fp32 -> bf16 convert (8 elems/thread) ----------------
__global__ __launch_bounds__(256)
void cvt_bf16_kernel(const float* __restrict__ in, bf16_t* __restrict__ out, int n) {
    const int i = (blockIdx.x * 256 + threadIdx.x) * 8;
    if (i + 7 < n) {
        const float4 a = *reinterpret_cast<const float4*>(in + i);
        const float4 c = *reinterpret_cast<const float4*>(in + i + 4);
        union { U4 u; bf16_t e[8]; } pk;
        pk.e[0] = (bf16_t)a.x; pk.e[1] = (bf16_t)a.y;
        pk.e[2] = (bf16_t)a.z; pk.e[3] = (bf16_t)a.w;
        pk.e[4] = (bf16_t)c.x; pk.e[5] = (bf16_t)c.y;
        pk.e[6] = (bf16_t)c.z; pk.e[7] = (bf16_t)c.w;
        *reinterpret_cast<U4*>(out + i) = pk.u;
    }
}

// ---------------- GEMM:  C[m,n] = sum_k A[m,k] * W[n,k] + bias[n] ----------------
// Double-buffered async global->LDS staging: next K-tile's copies are in
// flight (ASYNCcnt) while the current tile feeds the WMMA pipe.
// mode 0: bf16 row-major out        (Q, K projections)
// mode 1: bf16 transposed-for-V out (vT[b][h*64+d][s])
// mode 2: f32 out + residual        (final O projection)
__global__ __launch_bounds__(256)
void gemm_wt_kernel(const bf16_t* __restrict__ A, const bf16_t* __restrict__ W,
                    const float* __restrict__ bias, const float* __restrict__ resid,
                    bf16_t* __restrict__ outb, float* __restrict__ outf,
                    int N, int K, int mode) {
    __shared__ bf16_t As[2][128][40];
    __shared__ bf16_t Bs[2][128][40];
    constexpr unsigned kBufStride = sizeof(bf16_t) * 128 * 40;  // 10240 B

    const int tileM = blockIdx.y * 128;
    const int tileN = blockIdx.x * 128;
    const int t = threadIdx.x;
    const int lane = t & 31, waveId = t >> 5;
    const int wm = waveId & 1;       // 2 wave rows   (64 each)
    const int wn = waveId >> 1;      // 4 wave cols   (32 each)
    const int m  = lane & 15, hf = lane >> 4;

    v8f acc[4][2];
    #pragma unroll
    for (int i = 0; i < 4; ++i)
        #pragma unroll
        for (int j = 0; j < 2; ++j)
            acc[i][j] = (v8f){0.f, 0.f, 0.f, 0.f, 0.f, 0.f, 0.f, 0.f};

    const int lr = t >> 1;            // 0..127
    const int lc = (t & 1) << 4;      // 0 or 16
    const bf16_t* aG = A + (size_t)(tileM + lr) * K + lc;
    const bf16_t* wG = W + (size_t)(tileN + lr) * K + lc;
    const unsigned a0 = lds_addr32(&As[0][lr][lc]);
    const unsigned b0 = lds_addr32(&Bs[0][lr][lc]);

    auto issue_tile = [&](int buf, int kt) {
        const unsigned d = (unsigned)buf * kBufStride;
        async_ld_b128(a0 + d,      aG + kt);
        async_ld_b128(a0 + d + 16, aG + kt + 8);
        async_ld_b128(b0 + d,      wG + kt);
        async_ld_b128(b0 + d + 16, wG + kt + 8);
    };

    issue_tile(0, 0);
    int cur = 0;
    for (int kt = 0; kt < K; kt += 32) {
        wait_async0();                 // current tile's copies landed in LDS
        __syncthreads();               // visible to all waves; prev reads retired
        if (kt + 32 < K) {
            issue_tile(cur ^ 1, kt + 32);   // overlap next tile with compute
            __builtin_prefetch(aG + kt + 64, 0, 1);
            __builtin_prefetch(wG + kt + 64, 0, 1);
        }

        v16bf af[4], bw[2];
        #pragma unroll
        for (int i = 0; i < 4; ++i) {
            const bf16_t* p = &As[cur][wm * 64 + i * 16 + m][0];
            af[i] = load_frag2(p + hf * 16, p + hf * 16 + 8);
        }
        #pragma unroll
        for (int j = 0; j < 2; ++j) {
            const bf16_t* p = &Bs[cur][wn * 32 + j * 16 + m][0];
            bw[j] = load_frag2(p + hf * 16, p + hf * 16 + 8);
        }
        #pragma unroll
        for (int i = 0; i < 4; ++i)
            #pragma unroll
            for (int j = 0; j < 2; ++j)
                acc[i][j] = __builtin_amdgcn_wmma_f32_16x16x32_bf16(
                    false, af[i], false, bw[j], (short)0, acc[i][j], false, false);
        cur ^= 1;
    }

    #pragma unroll
    for (int i = 0; i < 4; ++i) {
        #pragma unroll
        for (int j = 0; j < 2; ++j) {
            #pragma unroll
            for (int r = 0; r < 8; ++r) {
                const int gm = tileM + wm * 64 + i * 16 + r + 8 * hf;
                const int gn = tileN + wn * 32 + j * 16 + m;
                float v = acc[i][j][r] + bias[gn];
                if (mode == 0) {
                    outb[(size_t)gm * N + gn] = (bf16_t)v;
                } else if (mode == 1) {
                    const int bb = gm >> 11;       // batch
                    const int sr = gm & (SEQ - 1); // seq pos
                    outb[((size_t)bb * D_MODEL + gn) * SEQ + sr] = (bf16_t)v;
                } else {
                    const size_t idx = (size_t)gm * N + gn;
                    outf[idx] = v + resid[idx];
                }
            }
        }
    }
}

// ---------------- Flash-style attention ----------------
// grid: (SEQ/64, N_HEADS, BATCH), block 128 (4 waves, 16 query rows each)
// K / V tiles double-buffered in LDS via async copies, shared by all 4 waves.
__global__ __launch_bounds__(128)
void attn_kernel(const bf16_t* __restrict__ q, const bf16_t* __restrict__ k,
                 const bf16_t* __restrict__ vT, bf16_t* __restrict__ ao) {
    const int t = threadIdx.x;
    const int wave = t >> 5, lane = t & 31;
    const int m = lane & 15, hf = lane >> 4;
    const int qb0 = blockIdx.x * 64 + wave * 16;
    const int h = blockIdx.y, b = blockIdx.z;
    const float scale = 0.125f; // 1/sqrt(64)

    __shared__ bf16_t P[4][16][40];    // per-wave P round-trip tile
    __shared__ bf16_t Ks[2][32][72];   // 32 keys x 64 dims, ping-pong
    __shared__ bf16_t Vs[2][64][40];   // 64 dims x 32 keys, ping-pong
    constexpr unsigned kKsStride = sizeof(bf16_t) * 32 * 72;  // 4608 B
    constexpr unsigned kVsStride = sizeof(bf16_t) * 64 * 40;  // 5120 B

    // Q A-fragments for d = 0..31 and d = 32..63 (read once)
    const bf16_t* qrow = q + (size_t)(b * SEQ + qb0 + m) * D_MODEL + h * D_K;
    const v16bf aq0 = load_frag2(qrow + hf * 16,      qrow + hf * 16 + 8);
    const v16bf aq1 = load_frag2(qrow + 32 + hf * 16, qrow + 40 + hf * 16);

    v8f o0, o1, o2, o3;
    o0 = o1 = o2 = o3 = (v8f){0.f, 0.f, 0.f, 0.f, 0.f, 0.f, 0.f, 0.f};
    float mrow[8], lrow[8];
    #pragma unroll
    for (int r = 0; r < 8; ++r) { mrow[r] = -1e30f; lrow[r] = 0.f; }

    // cooperative staging addresses (128 threads)
    const int kr = t >> 2, kc = (t & 3) << 4;   // K: 32 rows x 4 chunks of 16
    const int vr = t >> 1, vc = (t & 1) << 4;   // V: 64 rows x 2 chunks of 16
    const bf16_t* gk0 = k + (size_t)(b * SEQ + kr) * D_MODEL + h * D_K + kc;
    const bf16_t* gv0 = vT + ((size_t)(b * N_HEADS + h) * D_K + vr) * SEQ + vc;
    const unsigned ks0 = lds_addr32(&Ks[0][kr][kc]);
    const unsigned vs0 = lds_addr32(&Vs[0][vr][vc]);

    auto issue_tile = [&](int buf, int kb) {
        const unsigned dk = (unsigned)buf * kKsStride;
        const unsigned dv = (unsigned)buf * kVsStride;
        async_ld_b128(ks0 + dk,      gk0 + (size_t)kb * D_MODEL);
        async_ld_b128(ks0 + dk + 16, gk0 + (size_t)kb * D_MODEL + 8);
        async_ld_b128(vs0 + dv,      gv0 + kb);
        async_ld_b128(vs0 + dv + 16, gv0 + kb + 8);
    };

    issue_tile(0, 0);
    int cur = 0;
    for (int kb = 0; kb < SEQ; kb += 32) {
        wait_async0();                 // current K/V tiles landed
        __syncthreads();               // visible; previous tile reads retired
        if (kb + 32 < SEQ) issue_tile(cur ^ 1, kb + 32);

        // --- scores: S = Q * K^T, two 16-key half blocks ---
        v8f s0 = (v8f){0.f, 0.f, 0.f, 0.f, 0.f, 0.f, 0.f, 0.f};
        v8f s1 = s0;
        {
            const bf16_t* r0 = &Ks[cur][m][0];
            const bf16_t* r1 = &Ks[cur][16 + m][0];
            v16bf b00 = load_frag2(r0 + hf * 16,      r0 + hf * 16 + 8);
            v16bf b01 = load_frag2(r0 + 32 + hf * 16, r0 + 40 + hf * 16);
            s0 = __builtin_amdgcn_wmma_f32_16x16x32_bf16(false, aq0, false, b00, (short)0, s0, false, false);
            s0 = __builtin_amdgcn_wmma_f32_16x16x32_bf16(false, aq1, false, b01, (short)0, s0, false, false);
            v16bf b10 = load_frag2(r1 + hf * 16,      r1 + hf * 16 + 8);
            v16bf b11 = load_frag2(r1 + 32 + hf * 16, r1 + 40 + hf * 16);
            s1 = __builtin_amdgcn_wmma_f32_16x16x32_bf16(false, aq0, false, b10, (short)0, s1, false, false);
            s1 = __builtin_amdgcn_wmma_f32_16x16x32_bf16(false, aq1, false, b11, (short)0, s1, false, false);
        }

        // --- online softmax update (row = r + 8*hf, N across 16 lanes) ---
        #pragma unroll
        for (int r = 0; r < 8; ++r) {
            const float sm0 = s0[r] * scale;
            const float sm1 = s1[r] * scale;
            float mx = redmax16(fmaxf(sm0, sm1));
            const float mnew = fmaxf(mrow[r], mx);
            const float corr = __expf(mrow[r] - mnew);
            mrow[r] = mnew;
            const float p0 = __expf(sm0 - mnew);
            const float p1 = __expf(sm1 - mnew);
            lrow[r] = lrow[r] * corr + p0 + p1;
            o0[r] *= corr; o1[r] *= corr; o2[r] *= corr; o3[r] *= corr;
            P[wave][r + 8 * hf][m]      = (bf16_t)p0;
            P[wave][r + 8 * hf][16 + m] = (bf16_t)p1;
        }

        // --- re-load P in A-fragment layout (wave-local LDS round trip) ---
        const bf16_t* pr = &P[wave][m][0];
    const v16bf pa = load_frag2(pr + hf * 16, pr + hf * 16 + 8);

        // --- O += P * V  (contraction over 32 keys, 4 d-chunks of 16) ---
        const bf16_t* v0 = &Vs[cur][m][hf * 16];
        const bf16_t* v1 = &Vs[cur][16 + m][hf * 16];
        const bf16_t* v2 = &Vs[cur][32 + m][hf * 16];
        const bf16_t* v3 = &Vs[cur][48 + m][hf * 16];
        o0 = __builtin_amdgcn_wmma_f32_16x16x32_bf16(false, pa, false, load_frag2(v0, v0 + 8), (short)0, o0, false, false);
        o1 = __builtin_amdgcn_wmma_f32_16x16x32_bf16(false, pa, false, load_frag2(v1, v1 + 8), (short)0, o1, false, false);
        o2 = __builtin_amdgcn_wmma_f32_16x16x32_bf16(false, pa, false, load_frag2(v2, v2 + 8), (short)0, o2, false, false);
        o3 = __builtin_amdgcn_wmma_f32_16x16x32_bf16(false, pa, false, load_frag2(v3, v3 + 8), (short)0, o3, false, false);
        cur ^= 1;
    }

    // --- finalize: divide by row sums, store bf16 ---
    bf16_t* aoBase = ao + (size_t)(b * SEQ + qb0) * D_MODEL + h * D_K;
    #pragma unroll
    for (int r = 0; r < 8; ++r) {
        const float lsum = redsum16(lrow[r]);
        const float inv = 1.0f / lsum;
        const size_t rowOff = (size_t)(r + 8 * hf) * D_MODEL;
        aoBase[rowOff + m]      = (bf16_t)(o0[r] * inv);
        aoBase[rowOff + 16 + m] = (bf16_t)(o1[r] * inv);
        aoBase[rowOff + 32 + m] = (bf16_t)(o2[r] * inv);
        aoBase[rowOff + 48 + m] = (bf16_t)(o3[r] * inv);
    }
}

// ---------------- host launcher ----------------
extern "C" void kernel_launch(void* const* d_in, const int* in_sizes, int n_in,
                              void* d_out, int out_size, void* d_ws, size_t ws_size,
                              hipStream_t stream) {
    (void)in_sizes; (void)n_in; (void)out_size; (void)ws_size;
    const float* h   = (const float*)d_in[0];
    const float* Wq  = (const float*)d_in[1];
    const float* bq  = (const float*)d_in[2];
    const float* Wk  = (const float*)d_in[3];
    const float* bk  = (const float*)d_in[4];
    const float* Wv  = (const float*)d_in[5];
    const float* bv  = (const float*)d_in[6];
    const float* Wo  = (const float*)d_in[7];
    const float* bo  = (const float*)d_in[8];
    const float* lnw = (const float*)d_in[9];
    const float* lnb = (const float*)d_in[10];
    float* out = (float*)d_out;

    char* ws = (char*)d_ws;
    const size_t ROWS = (size_t)BATCH * SEQ;           // 4096
    const size_t MAT  = ROWS * D_MODEL;                // activations
    const size_t WMAT = (size_t)D_MODEL * D_MODEL;     // weights
    size_t off = 0;
    bf16_t* hn  = (bf16_t*)(ws + off); off += MAT  * 2;
    bf16_t* wqb = (bf16_t*)(ws + off); off += WMAT * 2;
    bf16_t* wkb = (bf16_t*)(ws + off); off += WMAT * 2;
    bf16_t* wvb = (bf16_t*)(ws + off); off += WMAT * 2;
    bf16_t* wob = (bf16_t*)(ws + off); off += WMAT * 2;
    bf16_t* qb  = (bf16_t*)(ws + off); off += MAT  * 2;
    bf16_t* kb  = (bf16_t*)(ws + off); off += MAT  * 2;
    bf16_t* vT  = (bf16_t*)(ws + off); off += MAT  * 2;
    bf16_t* ao  = (bf16_t*)(ws + off); off += MAT  * 2;

    ln_bf16_kernel<<<(unsigned)ROWS, 256, 0, stream>>>(h, lnw, lnb, hn);

    const int nW = (int)WMAT;
    cvt_bf16_kernel<<<nW / 2048, 256, 0, stream>>>(Wq, wqb, nW);
    cvt_bf16_kernel<<<nW / 2048, 256, 0, stream>>>(Wk, wkb, nW);
    cvt_bf16_kernel<<<nW / 2048, 256, 0, stream>>>(Wv, wvb, nW);
    cvt_bf16_kernel<<<nW / 2048, 256, 0, stream>>>(Wo, wob, nW);

    dim3 gGemm(D_MODEL / 128, (unsigned)(ROWS / 128));  // (8, 32)
    gemm_wt_kernel<<<gGemm, 256, 0, stream>>>(hn, wqb, bq, nullptr, qb, nullptr, D_MODEL, D_MODEL, 0);
    gemm_wt_kernel<<<gGemm, 256, 0, stream>>>(hn, wkb, bk, nullptr, kb, nullptr, D_MODEL, D_MODEL, 0);
    gemm_wt_kernel<<<gGemm, 256, 0, stream>>>(hn, wvb, bv, nullptr, vT, nullptr, D_MODEL, D_MODEL, 1);

    attn_kernel<<<dim3(SEQ / 64, N_HEADS, BATCH), 128, 0, stream>>>(qb, kb, vT, ao);

    gemm_wt_kernel<<<gGemm, 256, 0, stream>>>(ao, wob, bo, h, nullptr, out, D_MODEL, D_MODEL, 2);
}
